// VectorQuantizer_33603824124060
// MI455X (gfx1250) — matrix operand order repaired and checked by reading, not compiled, not run on hardware
//
#include <hip/hip_runtime.h>

// ---------------------------------------------------------------------------
// VQ argmin: indices[n] = argmin_k ( ||e_k||^2 - 2 * z_n . e_k )
// N=32768, K=8192, D=256.  bf16 hi/lo split GEMM on v_wmma_f32_16x16x32_bf16
// with fused running-argmin epilogue + next-chunk codebook prefetch.
// ---------------------------------------------------------------------------

typedef __attribute__((ext_vector_type(16))) __bf16 v16bf;
typedef __attribute__((ext_vector_type(8)))  float  v8f;

#define VQ_N   32768
#define VQ_K   8192
#define VQ_D   256
#define BM     128     // rows per block (8 waves x 16 rows)
#define BN     64      // codes per chunk (4 subtiles of 16)
#define NTHR   256

__device__ __forceinline__ void split_bf16(float x, __bf16& h, __bf16& l) {
    h = (__bf16)x;                 // RNE to bf16 (top ~8 mantissa bits)
    l = (__bf16)(x - (float)h);    // residual is exact in fp32; next 8 bits
}

__device__ __forceinline__ v8f wmma_bf16(v16bf a, v16bf b, v8f c) {
    return __builtin_amdgcn_wmma_f32_16x16x32_bf16(
        /*neg_a=*/false, a, /*neg_b=*/false, b,
        /*c_mod=*/(short)0, c, /*reuse_a=*/false, /*reuse_b=*/false);
}

// ---- kernel 1: ||e_k||^2 for all K codes (one wave per code) ---------------
__global__ void vq_esq_kernel(const float* __restrict__ cb, float* __restrict__ esq) {
    const int wave = threadIdx.x >> 5;
    const int lane = threadIdx.x & 31;
    const int code = blockIdx.x * 8 + wave;
    const float* r = cb + (size_t)code * VQ_D;
    float s = 0.0f;
#pragma unroll
    for (int i = 0; i < VQ_D / 32; ++i) {
        float v = r[lane + 32 * i];
        s = fmaf(v, v, s);
    }
#pragma unroll
    for (int m = 16; m >= 1; m >>= 1) s += __shfl_xor(s, m, 32);
    if (lane == 0) esq[code] = s;
}

// ---- kernel 2: main fused GEMM + argmin ------------------------------------
__global__ __launch_bounds__(NTHR, 1)
void vq_main_kernel(const float* __restrict__ z,
                    const float* __restrict__ cb,
                    const float* __restrict__ esq,
                    int* __restrict__ out) {
    // Codebook tile staged in LDS, already in B-fragment order:
    // index(sub, s, lane, e) = ((sub*8 + s)*32 + lane)*16 + e
    // where value = cb[chunk + sub*16 + (lane&15)][32*s + 16*(lane>>4) + e].
    __shared__ __align__(16) __bf16 lds_hi[BN * VQ_D];   // 32 KB
    __shared__ __align__(16) __bf16 lds_lo[BN * VQ_D];   // 32 KB

    const int tid  = threadIdx.x;
    const int lane = tid & 31;
    const int wave = tid >> 5;
    const int n15  = lane & 15;
    const int half = lane >> 4;
    const int rowbase = blockIdx.x * BM + wave * 16;
    const int row     = rowbase + n15;

    // ---- build persistent A fragments (z rows, hi+lo), contiguous-k order (same as B) -------
    // element e of slice s: d = 32*s + 16*half + e
    v16bf a_hi[8], a_lo[8];
    {
        const float* zr = z + (size_t)row * VQ_D;
#pragma unroll
        for (int s = 0; s < 8; ++s) {
            const int d0 = 32 * s + 16 * half;
            float4 f0 = *(const float4*)(zr + d0);
            float4 f1 = *(const float4*)(zr + d0 + 4);
            float4 f2 = *(const float4*)(zr + d0 + 8);
            float4 f3 = *(const float4*)(zr + d0 + 12);
            float v[16] = { f0.x, f0.y, f0.z, f0.w,  f1.x, f1.y, f1.z, f1.w,
                            f2.x, f2.y, f2.z, f2.w,  f3.x, f3.y, f3.z, f3.w };
#pragma unroll
            for (int e = 0; e < 16; ++e) {
                __bf16 h, l;
                split_bf16(v[e], h, l);
                a_hi[s][e] = h;
                a_lo[s][e] = l;
            }
        }
    }

    float minv[8];
    int   mini[8];
#pragma unroll
    for (int r = 0; r < 8; ++r) { minv[r] = 3.4e38f; mini[r] = 0; }

    // LDS-fill thread mapping (fixed per thread)
    const int ci  = tid >> 6;        // code sub-offset 0..3 within a 4-row step
    const int dq  = tid & 63;        // 64 threads cover one row of 256 dims
    const int d   = dq * 4;
    const int fs  = d >> 5;          // d-slice
    const int fh  = (d >> 4) & 1;    // half within slice
    const int fe  = d & 15;          // element base within half (0,4,8,12)

    for (int c = 0; c < VQ_K; c += BN) {
        __syncthreads();             // previous chunk's LDS reads done
        // ---- stage codebook chunk -> LDS (hi/lo bf16, frag-major) ---------
#pragma unroll
        for (int it = 0; it < BN / 4; ++it) {
            const int code = it * 4 + ci;              // 0..63 within chunk
            const float4 f = *(const float4*)(cb + (size_t)(c + code) * VQ_D + d);
            const int sub = code >> 4, n = code & 15;
            const int base = ((sub * 8 + fs) * 32 + fh * 16 + n) * 16 + fe;
            union { __bf16 b[4]; uint2 u; } ph, pl;
            split_bf16(f.x, ph.b[0], pl.b[0]);
            split_bf16(f.y, ph.b[1], pl.b[1]);
            split_bf16(f.z, ph.b[2], pl.b[2]);
            split_bf16(f.w, ph.b[3], pl.b[3]);
            *(uint2*)&lds_hi[base] = ph.u;
            *(uint2*)&lds_lo[base] = pl.u;
        }
        // ---- prefetch next chunk's codebook lines (global_prefetch_b8) ----
        if (c + BN < VQ_K) {
            const float* nxt = cb + (size_t)(c + BN) * VQ_D + tid * 64;
            __builtin_prefetch(nxt, 0, 3);        // first 128B line
            __builtin_prefetch(nxt + 32, 0, 3);   // second 128B line
        }
        __syncthreads();

        // ---- 4 independent 16x16 accumulation chains ----------------------
        v8f acc[4] = {};
#pragma unroll
        for (int s = 0; s < 8; ++s) {
#pragma unroll
            for (int sub = 0; sub < 4; ++sub) {
                const int bi = ((sub * 8 + s) * 32 + lane) * 16;
                const v16bf bh = *(const v16bf*)&lds_hi[bi];
                const v16bf bl = *(const v16bf*)&lds_lo[bi];
                acc[sub] = wmma_bf16(a_hi[s], bh, acc[sub]); // hi*hi
                acc[sub] = wmma_bf16(a_hi[s], bl, acc[sub]); // hi*lo
                acc[sub] = wmma_bf16(a_lo[s], bh, acc[sub]); // lo*hi
            }
        }

        // ---- epilogue: dist = esq - 2*cross; running argmin ---------------
#pragma unroll
        for (int sub = 0; sub < 4; ++sub) {
            const int code = c + sub * 16 + n15;
            const float es = esq[code];
#pragma unroll
            for (int r = 0; r < 8; ++r) {
                const float cand = fmaf(-2.0f, acc[sub][r], es);
                const bool lt = cand < minv[r];   // strict: keeps lowest index
                minv[r] = lt ? cand : minv[r];
                mini[r] = lt ? code : mini[r];
            }
        }
    }

    // ---- reduce across the 16 lanes holding each row's candidates ---------
#pragma unroll
    for (int m = 1; m <= 8; m <<= 1) {
#pragma unroll
        for (int r = 0; r < 8; ++r) {
            const float ov = __shfl_xor(minv[r], m, 32);
            const int   oi = __shfl_xor(mini[r], m, 32);
            const bool take = (ov < minv[r]) || (ov == minv[r] && oi < mini[r]);
            minv[r] = take ? ov : minv[r];
            mini[r] = take ? oi : mini[r];
        }
    }
    if (n15 == 0) {
#pragma unroll
        for (int r = 0; r < 8; ++r)
            out[rowbase + r + 8 * half] = mini[r];
    }
}

extern "C" void kernel_launch(void* const* d_in, const int* in_sizes, int n_in,
                              void* d_out, int out_size, void* d_ws, size_t ws_size,
                              hipStream_t stream) {
    const float* z   = (const float*)d_in[0];   // [N, D] fp32
    const float* cb  = (const float*)d_in[1];   // [K, D] fp32
    int*         out = (int*)d_out;             // [N] int32 indices
    float*       esq = (float*)d_ws;            // [K] fp32 scratch (32 KB)

    vq_esq_kernel<<<VQ_K / 8, NTHR, 0, stream>>>(cb, esq);
    vq_main_kernel<<<VQ_N / BM, NTHR, 0, stream>>>(z, cb, esq, out);
}
